// GRU_781684048417
// MI455X (gfx1250) — matrix operand order repaired and checked by reading, not compiled, not run on hardware
//
#include <hip/hip_runtime.h>
#include <hip/hip_bf16.h>
#include <cstdint>
#include <cstddef>

// ---------------------------------------------------------------------------
// Types for WMMA fragments (CDNA5 wave32: v_wmma_f32_16x16x32_f16)
// ---------------------------------------------------------------------------
typedef _Float16 v16h __attribute__((ext_vector_type(16)));
typedef _Float16 v8h  __attribute__((ext_vector_type(8)));
typedef float    v8f  __attribute__((ext_vector_type(8)));
typedef unsigned int u32x4 __attribute__((ext_vector_type(4)));
typedef int          i32x8 __attribute__((ext_vector_type(8)));
typedef int          i32x4 __attribute__((ext_vector_type(4)));

#if defined(__has_builtin)
#if __has_builtin(__builtin_amdgcn_tensor_load_to_lds) && \
    __has_builtin(__builtin_amdgcn_s_wait_tensorcnt)
#define GRU_USE_TDM 1
#endif
#endif
#ifndef GRU_USE_TDM
#define GRU_USE_TDM 0
#endif

// A-matrix 16x32 f16 fragment (matches B's contiguous order):
//   lane = half*16 + m ; element e -> K = half*16 + e
__device__ __forceinline__ v16h load_a_frag(const _Float16* __restrict__ A,
                                            int ld, int m, int half, int k0) {
  const _Float16* r = A + (size_t)m * ld + k0 + half * 16;
  v8h lo = *(const v8h*)(r);
  v8h hi = *(const v8h*)(r + 8);
  v16h a;
#pragma unroll
  for (int i = 0; i < 8; ++i) { a[i] = lo[i]; a[i + 8] = hi[i]; }
  return a;
}

// B-matrix 32x16 f16 fragment: lane = half*16 + n ; element e -> K = half*16+e.
// We compute h @ W^T with W row-major (N,K): B[k][n] = W[n][k], so each lane
// reads 16 contiguous f16 (32B) from row n of W. (Wrow = W + n*ldk)
__device__ __forceinline__ v16h load_b_frag(const _Float16* __restrict__ Wrow,
                                            int half, int k0) {
  return *(const v16h*)(Wrow + k0 + half * 16);
}

#if GRU_USE_TDM
// ---------------------------------------------------------------------------
// Tensor Data Mover: async DMA of one whh tile (nelem f16, 1-D) into LDS.
// D# per CDNA5 ISA:
//  group0: count=1 | lds_addr | global_addr[56:0] | type=2
//  group1: data_size=1 (2B), tensor_dim0=tile_dim0=nelem, tensor_dim1=tile_dim1=1
//  groups 2/3 (+aux): zero (<=2D tensor). Tracked by TENSORcnt.
// Issue from ONE wave only (TDM issues per-wave, EXEC ignored).
// This toolchain exposes the 6-arg builtin:
//   (u32x4 g0, i32x8 g1, i32x4 g2, i32x4 g3, i32x8 aux, i32 cpol)
// ---------------------------------------------------------------------------
__device__ __forceinline__ void tdm_load_whh(const _Float16* gsrc,
                                             unsigned lds_off, int nelem) {
  unsigned long long ga = (unsigned long long)(uintptr_t)gsrc;
  u32x4 g0;
  g0[0] = 1u;                                   // count=1, user mode
  g0[1] = lds_off;                              // LDS byte address
  g0[2] = (unsigned)ga;                         // global_addr[31:0]
  g0[3] = (unsigned)((ga >> 32) & 0x1FFFFFFu) | (2u << 30);  // [56:32] | type=2
  i32x8 g1;
  g1[0] = 0x00010000;                           // wg_mask=0, data_size=2B
  g1[1] = (int)(((unsigned)nelem & 0xFFFFu) << 16);          // tensor_dim0 lo
  g1[2] = (int)(((unsigned)nelem >> 16) | (1u << 16));       // td0 hi | td1=1
  g1[3] = (int)(((unsigned)nelem & 0xFFFFu) << 16);          // tile_dim0=nelem
  g1[4] = 1;                                    // tile_dim1=1, tile_dim2=0
  g1[5] = nelem;                                // tensor_dim0_stride lo
  g1[6] = 0;
  g1[7] = 0;
  i32x4 z4 = {0, 0, 0, 0};
  i32x8 z8 = {0, 0, 0, 0, 0, 0, 0, 0};
  __builtin_amdgcn_tensor_load_to_lds(g0, g1, z4, z4, z8, 0);
}
#endif

// ---------------------------------------------------------------------------
// fp32 -> fp16 weight conversion (grid-strided)
// ---------------------------------------------------------------------------
__global__ void f32_to_f16_kernel(const float* __restrict__ s,
                                  _Float16* __restrict__ d, int n) {
  for (int i = blockIdx.x * blockDim.x + threadIdx.x; i < n;
       i += gridDim.x * blockDim.x)
    d[i] = (_Float16)s[i];
}

// ---------------------------------------------------------------------------
// One full 96-layer GRU stack. Grid = 4 workgroups, each owns a 32-row batch
// slice for ALL layers/timesteps (batch is the only parallel recurrence dim).
// Block = 256 threads = 8 wave32.
//
// Phase A (per layer, time-parallel): xg[(b,t), 3H] = X_l @ wih_l^T + bih_l
// Phase B (per layer, sequential t): hg = h @ whh_l^T via WMMA on LDS-resident
//   whh; whh for layer l+1 is DMA'd by the TDM into the other LDS buffer while
//   layer l computes, retired with s_wait_tensorcnt at the layer boundary.
// ---------------------------------------------------------------------------
template <int H, int DIN0>
__global__ void __launch_bounds__(256) gru_stack_kernel(
    const float* __restrict__ x0,          // f32 layer-0 input (DIN0==3 path)
    const float* __restrict__ wih0_f32,    // (3H, DIN0) f32   (DIN0==3 path)
    const _Float16* __restrict__ xin,      // f16 layer-0 input, stride DIN0
    const _Float16* __restrict__ wih0_f16, // (3H, DIN0) f16
    const _Float16* __restrict__ wihL,     // (95, 3H, H) f16
    const _Float16* __restrict__ whhP,     // (96, 3H, H) f16 (slot0 = whh0)
    const float* __restrict__ bih0, const float* __restrict__ bihL,
    const float* __restrict__ bhh0, const float* __restrict__ bhhL,
    float* __restrict__ xg,                // (B*T, 3H) f32 scratch
    _Float16* __restrict__ out0,           // ping-pong activations, stride H
    _Float16* __restrict__ out1) {
  constexpr int T   = 96;
  constexpr int BSL = 32;        // batch rows per workgroup
  constexpr int G   = 3 * H;     // gate width
  constexpr int NT  = G / 16;    // N tiles
  constexpr int MT  = (BSL * T) / 16;  // 192 M tiles for phase A

  extern __shared__ char smem[];
  _Float16* s_whh0 = (_Float16*)smem;            // G*H f16 (weight buffer 0)
  _Float16* s_whh1 = s_whh0 + G * H;             // G*H f16 (weight buffer 1)
  _Float16* s_h    = s_whh1 + G * H;             // BSL*H f16 (hidden state)
  float*    s_hg   = (float*)(s_h + BSL * H);    // BSL*G f32 (recurrent gates)

  const int tid  = threadIdx.x;
  const int wave = tid >> 5;
  const int lane = tid & 31;
  const int half = lane >> 4;
  const int ln   = lane & 15;
  const size_t rowbase = (size_t)blockIdx.x * BSL * T;  // global (b*T+t) base

#if GRU_USE_TDM
  // Kick off the DMA of layer 0's whh before any compute.
  if (wave == 0)
    tdm_load_whh(whhP, (unsigned)(uintptr_t)s_whh0, G * H);
#endif

  for (int l = 0; l < 96; ++l) {
    const float* bih = (l == 0) ? bih0 : (bihL + (size_t)(l - 1) * G);

    // Warm next layer's wih (phase-A weights) toward the caches.
    if (l + 1 < 96)
      __builtin_prefetch(wihL + (size_t)l * G * H + (size_t)tid * 64, 0, 1);

    // ---------------- Phase A: input projection ----------------
    bool doneA = false;
    if constexpr (DIN0 % 32 != 0) {
      if (l == 0) {  // din = 3: trivial scalar projection
        const int total = BSL * T * G;
        for (int e = tid; e < total; e += 256) {
          int row = e / G, g = e - row * G;
          const float* xr = x0 + (rowbase + row) * DIN0;
          float acc = bih[g];
#pragma unroll
          for (int d = 0; d < DIN0; ++d) acc += xr[d] * wih0_f32[g * DIN0 + d];
          xg[(rowbase + row) * (size_t)G + g] = acc;
        }
        doneA = true;
      }
    }
    if (!doneA) {
      const bool l0   = (l == 0) && (DIN0 % 32 == 0);
      const int  K    = l0 ? DIN0 : H;
      const _Float16* W = l0 ? wih0_f16 : (wihL + (size_t)(l - 1) * G * H);
      const _Float16* X = (l == 0) ? xin : (((l - 1) & 1) ? out1 : out0);
      const int ldx = K;
      for (int tile = wave; tile < MT * NT; tile += 8) {
        const int mt = tile % MT, nt = tile / MT;
        v8f acc = {};
        const _Float16* Ab = X + (rowbase + (size_t)mt * 16) * ldx;
        const _Float16* Wr = W + (size_t)(nt * 16 + ln) * K;
        for (int k0 = 0; k0 < K; k0 += 32) {
          v16h a = load_a_frag(Ab, ldx, ln, half, k0);
          v16h b = load_b_frag(Wr, half, k0);
          acc = __builtin_amdgcn_wmma_f32_16x16x32_f16(
              false, a, false, b, (short)0, acc, false, false);
        }
        const float bv = bih[nt * 16 + ln];
        float* dst = xg + (rowbase + (size_t)mt * 16) * G + nt * 16 + ln;
#pragma unroll
        for (int r = 0; r < 8; ++r)
          dst[(size_t)(r + 8 * half) * G] = acc[r] + bv;
      }
    }

    // -------- Retire this layer's whh DMA; launch next layer's --------
#if GRU_USE_TDM
    if (wave == 0) {
      __builtin_amdgcn_s_wait_tensorcnt(0);  // whh for layer l is in LDS
      if (l + 1 < 96) {
        _Float16* nxt = ((l + 1) & 1) ? s_whh1 : s_whh0;
        tdm_load_whh(whhP + (size_t)(l + 1) * G * H,
                     (unsigned)(uintptr_t)nxt, G * H);
      }
    }
#else
    {
      const uint4* src = (const uint4*)(whhP + (size_t)l * G * H);
      uint4* dst = (uint4*)((l & 1) ? s_whh1 : s_whh0);
      const int n16 = G * H / 8;  // 16B chunks
      for (int i = tid; i < n16; i += 256) dst[i] = src[i];
    }
#endif
    for (int i = tid; i < BSL * H; i += 256) s_h[i] = (_Float16)0.f;
    __threadfence_block();
    __syncthreads();

    const _Float16* s_whh = (l & 1) ? s_whh1 : s_whh0;
    const float* bhh = (l == 0) ? bhh0 : (bhhL + (size_t)(l - 1) * G);
    _Float16* Y = (l & 1) ? out1 : out0;

    // ---------------- Phase B: sequential recurrence ----------------
    for (int t = 0; t < T; ++t) {
      // hg = h_prev @ whh^T  (M=32 -> 2 tiles, N=3H tiles, K=H)
      for (int tile = wave; tile < 2 * NT; tile += 8) {
        const int mt = tile & 1, nt = tile >> 1;
        v8f acc = {};
        const _Float16* Ab = s_h + mt * 16 * H;
        const _Float16* Wr = s_whh + (size_t)(nt * 16 + ln) * H;
#pragma unroll
        for (int k0 = 0; k0 < H; k0 += 32) {
          v16h a = load_a_frag(Ab, H, ln, half, k0);
          v16h b = load_b_frag(Wr, half, k0);
          acc = __builtin_amdgcn_wmma_f32_16x16x32_f16(
              false, a, false, b, (short)0, acc, false, false);
        }
        float* dst = s_hg + (size_t)(mt * 16) * G + nt * 16 + ln;
#pragma unroll
        for (int r = 0; r < 8; ++r) dst[(size_t)(r + 8 * half) * G] = acc[r];
      }
      __syncthreads();

      // Gate math + state update (sigmoid/tanh -> trans ops)
      for (int j = tid; j < BSL * H; j += 256) {
        const int bl = j / H, u = j - bl * H;
        const float* xgr = xg + (rowbase + (size_t)bl * T + t) * G;
        const float* hgr = s_hg + (size_t)bl * G;
        const float r_ = 1.f / (1.f + __expf(-(xgr[u]     + hgr[u]     + bhh[u])));
        const float z_ = 1.f / (1.f + __expf(-(xgr[H + u] + hgr[H + u] + bhh[H + u])));
        const float n_ = tanhf(xgr[2 * H + u] + r_ * (hgr[2 * H + u] + bhh[2 * H + u]));
        const float hp = (float)s_h[j];
        const float hn = (1.f - z_) * n_ + z_ * hp;
        s_h[j] = (_Float16)hn;
        Y[(rowbase + (size_t)bl * T + t) * H + u] = (_Float16)hn;
      }
      __syncthreads();
    }
    __threadfence_block();
    __syncthreads();
  }
}

// ---------------------------------------------------------------------------
// Dense head: relu(relu(relu(h@d1+b1)@d2+b2)@d3+b3), one block per batch row.
// ---------------------------------------------------------------------------
__global__ void __launch_bounds__(256) dense_head_kernel(
    const _Float16* __restrict__ y,  // (B, T, 64) f16, read t=95
    const float* __restrict__ w1, const float* __restrict__ b1,
    const float* __restrict__ w2, const float* __restrict__ b2,
    const float* __restrict__ w3, const float* __restrict__ b3,
    float* __restrict__ out) {
  __shared__ float h0[64], h1[64], h2[32];
  const int b = blockIdx.x, tid = threadIdx.x;
  if (tid < 64) h0[tid] = (float)y[((size_t)b * 96 + 95) * 64 + tid];
  __syncthreads();
  if (tid < 64) {
    float a = b1[tid];
#pragma unroll 8
    for (int k = 0; k < 64; ++k) a += h0[k] * w1[tid * 64 + k];
    h1[tid] = fmaxf(a, 0.f);
  }
  __syncthreads();
  if (tid < 32) {
    float a = b2[tid];
#pragma unroll 8
    for (int k = 0; k < 64; ++k) a += h1[k] * w2[tid * 64 + k];
    h2[tid] = fmaxf(a, 0.f);
  }
  __syncthreads();
  if (tid < 250) {
    float a = b3[tid];
#pragma unroll 8
    for (int k = 0; k < 32; ++k) a += h2[k] * w3[tid * 32 + k];
    out[(size_t)b * 250 + tid] = fmaxf(a, 0.f);
  }
}

// ---------------------------------------------------------------------------
// Host launcher
// ---------------------------------------------------------------------------
extern "C" void kernel_launch(void* const* d_in, const int* in_sizes, int n_in,
                              void* d_out, int out_size, void* d_ws,
                              size_t ws_size, hipStream_t stream) {
  (void)in_sizes; (void)n_in; (void)out_size; (void)ws_size;
  const float* x       = (const float*)d_in[0];
  const float* g1_wih0 = (const float*)d_in[1];
  const float* g1_bih0 = (const float*)d_in[2];
  const float* g1_whh0 = (const float*)d_in[3];
  const float* g1_bhh0 = (const float*)d_in[4];
  const float* g1_wih  = (const float*)d_in[5];
  const float* g1_bih  = (const float*)d_in[6];
  const float* g1_whh  = (const float*)d_in[7];
  const float* g1_bhh  = (const float*)d_in[8];
  const float* g2_wih0 = (const float*)d_in[9];
  const float* g2_bih0 = (const float*)d_in[10];
  const float* g2_whh0 = (const float*)d_in[11];
  const float* g2_bhh0 = (const float*)d_in[12];
  const float* g2_wih  = (const float*)d_in[13];
  const float* g2_bih  = (const float*)d_in[14];
  const float* g2_whh  = (const float*)d_in[15];
  const float* g2_bhh  = (const float*)d_in[16];
  const float* g3_wih0 = (const float*)d_in[17];
  const float* g3_bih0 = (const float*)d_in[18];
  const float* g3_whh0 = (const float*)d_in[19];
  const float* g3_bhh0 = (const float*)d_in[20];
  const float* g3_wih  = (const float*)d_in[21];
  const float* g3_bih  = (const float*)d_in[22];
  const float* g3_whh  = (const float*)d_in[23];
  const float* g3_bhh  = (const float*)d_in[24];
  const float* d1_w = (const float*)d_in[25];
  const float* d1_b = (const float*)d_in[26];
  const float* d2_w = (const float*)d_in[27];
  const float* d2_b = (const float*)d_in[28];
  const float* d3_w = (const float*)d_in[29];
  const float* d3_b = (const float*)d_in[30];

  // ---- carve workspace (256B aligned) ----
  size_t off = 0;
  auto carve = [&](size_t bytes) -> char* {
    char* p = (char*)d_ws + off;
    off += (bytes + 255) & ~(size_t)255;
    return p;
  };
  _Float16* ws_g1_wihL = (_Float16*)carve(95ull * 192 * 64 * 2);
  _Float16* ws_g1_whhP = (_Float16*)carve(96ull * 192 * 64 * 2);
  _Float16* ws_g2_wih0 = (_Float16*)carve(384ull * 64 * 2);
  _Float16* ws_g2_wihL = (_Float16*)carve(95ull * 384 * 128 * 2);
  _Float16* ws_g2_whhP = (_Float16*)carve(96ull * 384 * 128 * 2);
  _Float16* ws_g3_wih0 = (_Float16*)carve(192ull * 128 * 2);
  _Float16* ws_g3_wihL = (_Float16*)carve(95ull * 192 * 64 * 2);
  _Float16* ws_g3_whhP = (_Float16*)carve(96ull * 192 * 64 * 2);
  _Float16* ws_bufA = (_Float16*)carve(128ull * 96 * 128 * 2);
  _Float16* ws_bufB = (_Float16*)carve(128ull * 96 * 128 * 2);
  _Float16* ws_bufC = (_Float16*)carve(128ull * 96 * 128 * 2);
  float*    ws_xg   = (float*)carve(128ull * 96 * 384 * 4);

  auto cvt = [&](const float* s, _Float16* d, int n) {
    int blocks = (n + 255) / 256;
    if (blocks > 1024) blocks = 1024;
    f32_to_f16_kernel<<<blocks, 256, 0, stream>>>(s, d, n);
  };
  // g1 (wih0 stays f32 for the din=3 scalar path)
  cvt(g1_wih, ws_g1_wihL, 95 * 192 * 64);
  cvt(g1_whh0, ws_g1_whhP, 192 * 64);
  cvt(g1_whh, ws_g1_whhP + 192 * 64, 95 * 192 * 64);
  // g2
  cvt(g2_wih0, ws_g2_wih0, 384 * 64);
  cvt(g2_wih, ws_g2_wihL, 95 * 384 * 128);
  cvt(g2_whh0, ws_g2_whhP, 384 * 128);
  cvt(g2_whh, ws_g2_whhP + 384 * 128, 95 * 384 * 128);
  // g3
  cvt(g3_wih0, ws_g3_wih0, 192 * 128);
  cvt(g3_wih, ws_g3_wihL, 95 * 192 * 64);
  cvt(g3_whh0, ws_g3_whhP, 192 * 64);
  cvt(g3_whh, ws_g3_whhP + 192 * 64, 95 * 192 * 64);

  // LDS: 2x whh(G*H f16) + h(32*H f16) + hg(32*G f32)
  const size_t lds64  = 2 * 3 * 64 * 64 * 2 + 32 * 64 * 2 + 32 * 192 * 4;    //  76KB
  const size_t lds128 = 2 * 3 * 128 * 128 * 2 + 32 * 128 * 2 + 32 * 384 * 4; // 248KB
  (void)hipFuncSetAttribute((const void*)&gru_stack_kernel<128, 64>,
                            hipFuncAttributeMaxDynamicSharedMemorySize,
                            (int)lds128);
  (void)hipFuncSetAttribute((const void*)&gru_stack_kernel<64, 3>,
                            hipFuncAttributeMaxDynamicSharedMemorySize,
                            (int)lds64);
  (void)hipFuncSetAttribute((const void*)&gru_stack_kernel<64, 128>,
                            hipFuncAttributeMaxDynamicSharedMemorySize,
                            (int)lds64);

  // g1: din=3, h=64. out in bufB (layer 95 -> out1).
  gru_stack_kernel<64, 3><<<4, 256, lds64, stream>>>(
      x, g1_wih0, nullptr, nullptr, ws_g1_wihL, ws_g1_whhP,
      g1_bih0, g1_bih, g1_bhh0, g1_bhh, ws_xg, ws_bufA, ws_bufB);
  // g2: din=64, h=128. in bufB, out in bufC.
  gru_stack_kernel<128, 64><<<4, 256, lds128, stream>>>(
      nullptr, nullptr, ws_bufB, ws_g2_wih0, ws_g2_wihL, ws_g2_whhP,
      g2_bih0, g2_bih, g2_bhh0, g2_bhh, ws_xg, ws_bufA, ws_bufC);
  // g3: din=128, h=64. in bufC, out in bufB.
  gru_stack_kernel<64, 128><<<4, 256, lds64, stream>>>(
      nullptr, nullptr, ws_bufC, ws_g3_wih0, ws_g3_wihL, ws_g3_whhP,
      g3_bih0, g3_bih, g3_bhh0, g3_bhh, ws_xg, ws_bufA, ws_bufB);

  dense_head_kernel<<<128, 256, 0, stream>>>(
      ws_bufB, d1_w, d1_b, d2_w, d2_b, d3_w, d3_b, (float*)d_out);
}
